// MFA_86492051407161
// MI455X (gfx1250) — matrix operand order repaired and checked by reading, not compiled, not run on hardware
//
#include <hip/hip_runtime.h>
#include <hip/hip_bf16.h>
#include <math.h>

typedef __attribute__((ext_vector_type(16))) __bf16  v16bf;
typedef __attribute__((ext_vector_type(8)))  __bf16  v8bf;
typedef __attribute__((ext_vector_type(8)))  float   v8f;

#define MFA_D   128
#define MFA_K   32
#define MFA_L   4
#define NCOLS   192            // 32 (quad) + 32 (cross) + 128 (y = K*l)
#define WAVES_PER_BLOCK 4
#define WSTRIDE 136            // padded LDS column stride (bf16 elems): 272B -> bank-balanced

// workspace layout (bytes)
#define WS_WH_OFF   0
#define WS_WL_OFF   (NCOLS*MFA_D*2)                 // 49152
#define WS_IL_OFF   (WS_WL_OFF + NCOLS*MFA_D*2)    // 98304
#define WS_C_OFF    (WS_IL_OFF + MFA_K*16*4)       // 100352
#define WS_MB_OFF   (WS_C_OFF + MFA_K*4)           // 100480

// ---------------------------------------------------------------------------
// Phase 1: build fused weight matrix W[192][128] split into bf16 hi/lo,
// invert Woodbury 4x4 per component, fold constants. One wave, k = component.
// ---------------------------------------------------------------------------
__global__ void mfa_prep(const float* __restrict__ MU, const float* __restrict__ A,
                         const float* __restrict__ D, const float* __restrict__ PI,
                         unsigned short* __restrict__ WhU, unsigned short* __restrict__ WlU,
                         float* __restrict__ iLg, float* __restrict__ Cg,
                         float* __restrict__ mbg)
{
    const int k = threadIdx.x;
    if (k >= MFA_K) return;
    __bf16* Wh = (__bf16*)WhU;
    __bf16* Wl = (__bf16*)WlU;

    float Lm[4][4], inv[4][4];
#pragma unroll
    for (int i = 0; i < 4; ++i)
#pragma unroll
        for (int j = 0; j < 4; ++j) { Lm[i][j] = (i == j) ? 1.f : 0.f;
                                      inv[i][j] = (i == j) ? 1.f : 0.f; }
    float mb[4] = {0.f, 0.f, 0.f, 0.f};
    float tc = 0.f, sld = 0.f;

    for (int dd = 0; dd < MFA_D; ++dd) {
        float Dv  = D[k*MFA_D + dd];
        float dsq = Dv * Dv;
        float id  = 1.0f / dsq;
        float mu  = MU[k*MFA_D + dd];
        tc  += id * mu * mu;
        sld += logf(dsq);
        {   // column k: iD   (paired with x^2 operand -> t_quad)
            float v = id;
            __bf16 h = (__bf16)v; __bf16 l = (__bf16)(v - (float)h);
            Wh[k*MFA_D + dd] = h;  Wl[k*MFA_D + dd] = l;
        }
        {   // column 32+k: iD*MU  (-> t_cross)
            float v = id * mu;
            __bf16 h = (__bf16)v; __bf16 l = (__bf16)(v - (float)h);
            Wh[(32 + k)*MFA_D + dd] = h;  Wl[(32 + k)*MFA_D + dd] = l;
        }
        float a[4], b[4];
#pragma unroll
        for (int li = 0; li < 4; ++li) {
            a[li] = A[(k*MFA_D + dd)*MFA_L + li];
            b[li] = id * a[li];
            const int col = 64 + 4*k + li;          // columns 64..191: B
            __bf16 h = (__bf16)b[li]; __bf16 l = (__bf16)(b[li] - (float)h);
            Wh[col*MFA_D + dd] = h;  Wl[col*MFA_D + dd] = l;
            mb[li] += mu * b[li];
        }
#pragma unroll
        for (int i = 0; i < 4; ++i)
#pragma unroll
            for (int j = 0; j < 4; ++j) Lm[i][j] += a[i] * b[j];
    }

    // Gauss-Jordan inverse (L is SPD: I + A^T iD A, no pivoting needed)
    float det = 1.f;
#pragma unroll
    for (int i = 0; i < 4; ++i) {
        float p = Lm[i][i];
        det *= p;
        float ip = 1.f / p;
#pragma unroll
        for (int j = 0; j < 4; ++j) { Lm[i][j] *= ip; inv[i][j] *= ip; }
#pragma unroll
        for (int r = 0; r < 4; ++r) {
            if (r == i) continue;
            float f = Lm[r][i];
#pragma unroll
            for (int j = 0; j < 4; ++j) { Lm[r][j] -= f*Lm[i][j]; inv[r][j] -= f*inv[i][j]; }
        }
    }

#pragma unroll
    for (int i = 0; i < 4; ++i)
#pragma unroll
        for (int j = 0; j < 4; ++j) iLg[k*16 + i*4 + j] = inv[i][j];
#pragma unroll
    for (int li = 0; li < 4; ++li) mbg[k*4 + li] = mb[li];

    const float LOG2PI = 1.8378770664093453f;
    // NOTE: faithful to the source, which adds PI (not log PI).
    Cg[k] = PI[k] - 0.5f*((float)MFA_D*LOG2PI + logf(det) + sld + tc);
}

// ---------------------------------------------------------------------------
// 16x16 D-tile via 4 K-steps x 3 split-products of v_wmma_f32_16x16x32_bf16.
// B operands come from the LDS-resident padded weight table (ds_load_b128 x2).
// ---------------------------------------------------------------------------
__device__ __forceinline__ v8f mfa_gemm_tile(const __bf16* sWh, const __bf16* sWl,
                                             const v16bf* Ah, const v16bf* Al,
                                             int col, int half)
{
    v8f acc = {0.f,0.f,0.f,0.f,0.f,0.f,0.f,0.f};
#pragma unroll
    for (int kt = 0; kt < 4; ++kt) {
        // B layout (32x16, bf16): lane n%16 = column, elem e<8 -> K = kt*32 + 8*(lane/16) + e, e>=8 -> K = kt*32 + 16 + 8*(lane/16) + (e-8)
        const int off = col*WSTRIDE + kt*32 + half*8;    // 16-B aligned (272B stride)
        union { v16bf v; v8bf h[2]; } bh, bl;
        bh.h[0] = *(const v8bf*)(sWh + off);
        bh.h[1] = *(const v8bf*)(sWh + off + 16);
        bl.h[0] = *(const v8bf*)(sWl + off);
        bl.h[1] = *(const v8bf*)(sWl + off + 16);
        acc = __builtin_amdgcn_wmma_f32_16x16x32_bf16(false, Ah[kt], false, bh.v, (short)0, acc, false, false);
        acc = __builtin_amdgcn_wmma_f32_16x16x32_bf16(false, Ah[kt], false, bl.v, (short)0, acc, false, false);
        acc = __builtin_amdgcn_wmma_f32_16x16x32_bf16(false, Al[kt], false, bh.v, (short)0, acc, false, false);
    }
    return acc;
}

// ---------------------------------------------------------------------------
// Phase 2: one wave per 16-row M-tile. W staged in LDS once per block;
// GEMM -> per-wave LDS scratch -> transpose epilogue + logsumexp.
// ---------------------------------------------------------------------------
__global__ __launch_bounds__(WAVES_PER_BLOCK*32) void mfa_main(
    const float* __restrict__ x,
    const unsigned short* __restrict__ WhU,
    const unsigned short* __restrict__ WlU,
    const float* __restrict__ iLg, const float* __restrict__ Cg,
    const float* __restrict__ mbg,
    float* __restrict__ out, int N, int numTiles)
{
    __shared__ __bf16 s_Wh[NCOLS*WSTRIDE];              // 52.2 KB
    __shared__ __bf16 s_Wl[NCOLS*WSTRIDE];              // 52.2 KB
    __shared__ float  sscr[WAVES_PER_BLOCK][NCOLS*16];  // 48 KB per-wave D-tile scratch
    __shared__ float  s_iL[MFA_K*16];
    __shared__ float  s_mb[MFA_K*4];
    __shared__ float  s_C[MFA_K];

    const int tid = threadIdx.x;

    // ---- stage weight table into padded LDS (once per block) ----
    for (int idx = tid; idx < NCOLS*16; idx += WAVES_PER_BLOCK*32) {
        const int col = idx >> 4;
        const int q   = idx & 15;                       // 16-byte chunk within column
        ((uint4*)(s_Wh + col*WSTRIDE))[q] = ((const uint4*)(WhU + col*MFA_D))[q];
        ((uint4*)(s_Wl + col*WSTRIDE))[q] = ((const uint4*)(WlU + col*MFA_D))[q];
    }
    for (int i = tid; i < MFA_K*16; i += WAVES_PER_BLOCK*32) s_iL[i] = iLg[i];
    if (tid < MFA_K*4) s_mb[tid] = mbg[tid];
    if (tid < MFA_K)   s_C[tid]  = Cg[tid];
    __syncthreads();

    const int wslot = tid >> 5;
    const int lane  = tid & 31;
    const int half  = lane >> 4;   // which 16-lane half
    const int mrow  = lane & 15;
    float* sp = sscr[wslot];
    const int step = (int)gridDim.x * WAVES_PER_BLOCK;

    for (int t0 = (int)blockIdx.x * WAVES_PER_BLOCK; t0 < numTiles; t0 += step) {
        const int tile = t0 + wslot;
        const bool active = (tile < numTiles);   // wave-uniform

        v16bf axh[4], axl[4], ash[4], asl[4];
        if (active) {
            int row = tile*16 + mrow;
            if (row >= N) row = N - 1;
            const float* xr = x + (size_t)row * MFA_D;

            // prefetch next tile's x rows (global_prefetch_b8)
            const int nt2 = tile + step;
            if (nt2 < numTiles)
                __builtin_prefetch(x + (size_t)(nt2*16 + mrow) * MFA_D, 0, 0);

            // Build A operands in ISA 16-bit 16x32 layout:
            // lanes 0-15:  M=lane,    elems 0..7 -> K=kt*32+0..7,  8..15 -> K=kt*32+16..23
            // lanes 16-31: M=lane-16, elems 0..7 -> K=kt*32+8..15, 8..15 -> K=kt*32+24..31
#pragma unroll
            for (int kt = 0; kt < 4; ++kt) {
#pragma unroll
                for (int c = 0; c < 2; ++c) {
                    const int kb = kt*32 + half*8 + c*16;
#pragma unroll
                    for (int j = 0; j < 8; ++j) {
                        float v = xr[kb + j];
                        __bf16 h = (__bf16)v;
                        __bf16 l = (__bf16)(v - (float)h);
                        float sq = v * v;
                        __bf16 sh = (__bf16)sq;
                        __bf16 sl = (__bf16)(sq - (float)sh);
                        const int e = c*8 + j;
                        axh[kt][e] = h;  axl[kt][e] = l;
                        ash[kt][e] = sh; asl[kt][e] = sl;
                    }
                }
            }

            // cols 0..31: t_quad (A = x^2)
            for (int nt = 0; nt < 2; ++nt) {
                const int col = nt*16 + mrow;
                v8f acc = mfa_gemm_tile(s_Wh, s_Wl, ash, asl, col, half);
                float* dphase = sp + col*16 + half*8;   // D layout: VGPR r -> M = r + 8*half
#pragma unroll
                for (int r = 0; r < 8; ++r) dphase[r] = acc[r];
            }
            // cols 32..191: t_cross + y (A = x)
            for (int nt = 2; nt < 12; ++nt) {
                const int col = nt*16 + mrow;
                v8f acc = mfa_gemm_tile(s_Wh, s_Wl, axh, axl, col, half);
                float* dphase = sp + col*16 + half*8;
#pragma unroll
                for (int r = 0; r < 8; ++r) dphase[r] = acc[r];
            }
        }
        __syncthreads();

        if (active) {
            // lane (mrow, half) handles row=mrow, components k = half*16 .. half*16+15
            float ll[16];
            float m = -3.402823e38f;
#pragma unroll
            for (int kk = 0; kk < 16; ++kk) {
                const int k = half*16 + kk;
                float q  = sp[k*16 + mrow];
                float cc = sp[(32 + k)*16 + mrow];
                float y0 = sp[(64 + 4*k + 0)*16 + mrow] - s_mb[k*4 + 0];
                float y1 = sp[(64 + 4*k + 1)*16 + mrow] - s_mb[k*4 + 1];
                float y2 = sp[(64 + 4*k + 2)*16 + mrow] - s_mb[k*4 + 2];
                float y3 = sp[(64 + 4*k + 3)*16 + mrow] - s_mb[k*4 + 3];
                const float* il = s_iL + k*16;
                float t2 = y0*(il[0]*y0  + il[1]*y1  + il[2]*y2  + il[3]*y3)
                         + y1*(il[4]*y0  + il[5]*y1  + il[6]*y2  + il[7]*y3)
                         + y2*(il[8]*y0  + il[9]*y1  + il[10]*y2 + il[11]*y3)
                         + y3*(il[12]*y0 + il[13]*y1 + il[14]*y2 + il[15]*y3);
                float v = s_C[k] + cc - 0.5f*q + 0.5f*t2;
                ll[kk] = v;
                m = fmaxf(m, v);
            }
            float s = 0.f;
#pragma unroll
            for (int kk = 0; kk < 16; ++kk) s += expf(ll[kk] - m);
            // combine the two k-halves across lane pairs (wave32 xor-16)
            float m2 = __shfl_xor(m, 16, 32);
            float s2 = __shfl_xor(s, 16, 32);
            float M  = fmaxf(m, m2);
            float S  = expf(m - M)*s + expf(m2 - M)*s2;
            if (half == 0) {
                const int row = tile*16 + mrow;
                if (row < N) out[row] = M + logf(S);
            }
        }
        __syncthreads();
    }
}

// ---------------------------------------------------------------------------
extern "C" void kernel_launch(void* const* d_in, const int* in_sizes, int n_in,
                              void* d_out, int out_size, void* d_ws, size_t ws_size,
                              hipStream_t stream)
{
    const float* x  = (const float*)d_in[0];
    const float* MU = (const float*)d_in[1];
    const float* A  = (const float*)d_in[2];
    const float* D  = (const float*)d_in[3];
    const float* PI = (const float*)d_in[4];
    float* out = (float*)d_out;
    const int N = in_sizes[0] / MFA_D;

    char* ws = (char*)d_ws;
    unsigned short* WhU = (unsigned short*)(ws + WS_WH_OFF);
    unsigned short* WlU = (unsigned short*)(ws + WS_WL_OFF);
    float* iLg = (float*)(ws + WS_IL_OFF);
    float* Cg  = (float*)(ws + WS_C_OFF);
    float* mbg = (float*)(ws + WS_MB_OFF);

    mfa_prep<<<1, 32, 0, stream>>>(MU, A, D, PI, WhU, WlU, iLg, Cg, mbg);

    const int numTiles = (N + 15) / 16;
    int blocks = (numTiles + WAVES_PER_BLOCK - 1) / WAVES_PER_BLOCK;
    if (blocks > 1024) blocks = 1024;
    mfa_main<<<blocks, WAVES_PER_BLOCK*32, 0, stream>>>(
        x, WhU, WlU, iLg, Cg, mbg, out, N, numTiles);
}
